// UpdateNode_85744727097792
// MI455X (gfx1250) — matrix operand order repaired and checked by reading, not compiled, not run on hardware
//
#include <hip/hip_runtime.h>
#include <hip/hip_bf16.h>
#include <math.h>

#define NNODES 25000
#define NEDGES 150000
#define EPSV 1e-8f

// sqrt(1/95), sqrt(3/95), sqrt(5/95)
#define A0C 0.10259783520851541f
#define A1C 0.17770466332544357f
#define A2C 0.22941573387056174f
#define C_OLD 0.8944271909999159f
#define C_NEW 0.4472135954999579f
#define NORM8 0.35355339059327373f

typedef __bf16 bf16_t;
typedef __attribute__((ext_vector_type(16))) __bf16 v16bf;
typedef __attribute__((ext_vector_type(8)))  __bf16 v8bf;
typedef __attribute__((ext_vector_type(8)))  float  v8f;

// ---- packed bf16 weight-fragment offsets (elements) ----
// layout per matrix (K x N): tile (kb,nb) at ((kb*ntiles_n + nb)*32 + lane)*16 + i
// h=lane/16: elem i<8 = W[kb*32+8h+i][nb*16+lane%16]; elem i>=8 = W[kb*32+16+8h+i-8][nb*16+lane%16]
#define OFF_WM0   0u        // 288x320
#define OFF_WM1R  92160u    // 192x128
#define OFF_WM1I  116736u
#define OFF_WM1IN 141312u   // -W_m1i
#define OFF_WM2R  165888u   // 96x64
#define OFF_WM2I  172032u
#define OFF_WM2IN 178176u   // -W_m2i
#define OFF_WRAD  184320u   // 64x320
#define OFF_WENV  204800u   // 64x192
#define OFF_WP0   217088u   // 64x64
#define OFF_WP1   221184u
#define OFF_WP2   225280u
#define WPACK_TOTAL 229376u

__device__ __forceinline__ v8f wmma_bf16(v16bf a, v16bf b, v8f c) {
  return __builtin_amdgcn_wmma_f32_16x16x32_bf16(false, a, false, b, (short)0, c, false, false);
}

// A-fragment (16x32 bf16) from LDS, row stride ld, K-tile base column c0.
// Both halves are 16B-aligned by construction -> two ds_load_b128.
__device__ __forceinline__ v16bf afrag_lds(const bf16_t* X, int ld, int c0, int lane) {
  int row = lane & 15;
  const bf16_t* p = X + row * ld + c0 + ((lane & 16) ? 8 : 0);
  v8bf lo = *(const v8bf*)(p);
  v8bf hi = *(const v8bf*)(p + 16);
  return __builtin_shufflevector(lo, hi, 0, 1, 2, 3, 4, 5, 6, 7, 8, 9, 10, 11, 12, 13, 14, 15);
}

// B-fragment (32x16 bf16): one contiguous 32B vector per lane.
__device__ __forceinline__ v16bf bfrag(const bf16_t* tile_base, int lane) {
  return *(const v16bf*)(tile_base + lane * 16);
}

// store 16x16 f32 D tile to LDS, row = edge, col = output channel
__device__ __forceinline__ void store_tile(float* Y, int ld, int col0, int lane, v8f d) {
  int col = col0 + (lane & 15);
  int r0 = (lane & 16) ? 8 : 0;
#pragma unroll
  for (int r = 0; r < 8; ++r) Y[(r0 + r) * ld + col] = d[r];
}

__device__ __forceinline__ void atomic_add_f32(float* p, float v) {
  __hip_atomic_fetch_add(p, v, __ATOMIC_RELAXED, __HIP_MEMORY_SCOPE_AGENT);
}

// ---------------- weight packing: f32 (K x N) -> bf16 fragment order ----------------
__global__ void pack_weights_kernel(const float* __restrict__ W, bf16_t* __restrict__ out,
                                    int K, int Nn, float scale) {
  int idx = blockIdx.x * 256 + threadIdx.x;
  if (idx >= K * Nn) return;
  int i    = idx & 15;
  int lane = (idx >> 4) & 31;
  int tile = idx >> 9;
  int ntn  = Nn >> 4;
  int nb = tile % ntn, kb = tile / ntn;
  int k = kb * 32 + ((i < 8) ? ((lane >> 4) * 8 + i) : (16 + (lane >> 4) * 8 + (i - 8)));
  int n = nb * 16 + (lane & 15);
  out[idx] = (bf16_t)(W[(size_t)k * Nn + n] * scale);
}

// ---------------- per-node layernorm (one wave32 per node) ----------------
__global__ void node_ln_kernel(const float* __restrict__ nf, float* __restrict__ nn,
                               const float* __restrict__ ln_w0, const float* __restrict__ ln_b0,
                               const float* __restrict__ ln_w1, const float* __restrict__ ln_w2) {
  int node = blockIdx.x * 8 + (threadIdx.x >> 5);
  int lane = threadIdx.x & 31;
  if (node >= NNODES) return;
  const float* x = nf + (size_t)node * 576;
  float a = x[lane], b = x[32 + lane];
  float s = a + b, ss = a * a + b * b;
  float s1 = 0.f, s2 = 0.f;
  for (int j = lane; j < 192; j += 32) { float v = x[64 + j];  s1 += v * v; }
  for (int j = lane; j < 320; j += 32) { float v = x[256 + j]; s2 += v * v; }
#pragma unroll
  for (int m = 16; m >= 1; m >>= 1) {
    s  += __shfl_xor(s,  m, 32);
    ss += __shfl_xor(ss, m, 32);
    s1 += __shfl_xor(s1, m, 32);
    s2 += __shfl_xor(s2, m, 32);
  }
  float mu = s * (1.f / 64.f);
  float var = ss * (1.f / 64.f) - mu * mu;
  float inv0 = rsqrtf(var + EPSV);
  float vL = 0.5f * (s1 * (1.f / 192.f) + s2 * (1.f / 320.f));
  float invL = rsqrtf(vL + EPSV);
  float* o = nn + (size_t)node * 576;
  o[lane]      = (x[lane]      - mu) * inv0 * ln_w0[lane]      + ln_b0[lane];
  o[32 + lane] = (x[32 + lane] - mu) * inv0 * ln_w0[32 + lane] + ln_b0[32 + lane];
  for (int j = lane; j < 192; j += 32) { int c = j / 3; o[64 + j]  = x[64 + j]  * invL * ln_w1[c]; }
  for (int j = lane; j < 320; j += 32) { int c = j / 5; o[256 + j] = x[256 + j] * invL * ln_w2[c]; }
}

// ---------------- main edge kernel: one wave32 per 16-edge tile ----------------
__launch_bounds__(32)
__global__ void edge_kernel(const float* __restrict__ nn,
                            float* __restrict__ accg,
                            const float* __restrict__ latents,
                            const float* __restrict__ edge_features,
                            const float* __restrict__ wigner,
                            const int*  __restrict__ edge_src,
                            const float* __restrict__ b_rad,
                            const float* __restrict__ bp0,
                            const bf16_t* __restrict__ wpack) {
  extern __shared__ char smem[];
  bf16_t* XR    = (bf16_t*)(smem);            // 9*16*96 bf16 = 27648B (reused as MS 9*16*64 bf16)
  bf16_t* latbf = (bf16_t*)(smem + 27648);    // 16*64 bf16 = 2048B
  float*  Y0S   = (float*)(smem + 29696);     // 16*320 = 20480B
  float*  Yp1S  = (float*)(smem + 50176);     // 16*128 = 8192B
  float*  Ym1S  = (float*)(smem + 58368);     // 8192B
  float*  Yp2S  = (float*)(smem + 66560);     // 16*64 = 4096B
  float*  Ym2S  = (float*)(smem + 70656);     // 4096B
  float*  radS  = (float*)(smem + 74752);     // 16*320 = 20480B
  float*  wS    = (float*)(smem + 95232);     // 16*192 = 12288B
  float*  D0S   = (float*)(smem + 107520);    // 16
  float*  D1S   = (float*)(smem + 107584);    // 16*9
  float*  D2S   = (float*)(smem + 108160);    // 16*25
  int*    ecS   = (int*)  (smem + 109760);    // 16
  // total 109824 B dynamic LDS

  const int lane = threadIdx.x;
  const int e0 = blockIdx.x * 16;

  // ---- phase 0: stage lat, wigner blocks, source nodes ----
  __builtin_prefetch(wpack + OFF_WM0 + lane * 64, 0, 0);
  if (lane < 16) ecS[lane] = edge_src[e0 + lane];
  for (int j = lane; j < 16 * 64; j += 32) {
    int e = j >> 6, c = j & 63;
    latbf[j] = (bf16_t)latents[(size_t)(e0 + e) * 64 + c];
  }
  for (int j = lane; j < 16 * 35; j += 32) {
    int e = j / 35, k = j % 35;
    const float* wg = wigner + (size_t)(e0 + e) * 81;
    if (k == 0) D0S[e] = wg[0];
    else if (k < 10) { int kk = k - 1;  D1S[e * 9 + kk]  = wg[(1 + kk / 3) * 9 + (1 + kk % 3)]; }
    else             { int kk = k - 10; D2S[e * 25 + kk] = wg[(4 + kk / 5) * 9 + (4 + kk % 5)]; }
  }
  __syncthreads();

  const int colL = lane & 15;
  const int r0 = (lane & 16) ? 8 : 0;

  // ---- phase 1: rad = lat@W_rad + b_rad ; w = lat@W_env ----
  {
    v16bf al[2];
#pragma unroll
    for (int kb = 0; kb < 2; ++kb) al[kb] = afrag_lds(latbf, 64, kb * 32, lane);
    for (int nb = 0; nb < 20; ++nb) {
      v8f acc = {};
#pragma unroll
      for (int kb = 0; kb < 2; ++kb)
        acc = wmma_bf16(al[kb], bfrag(wpack + OFF_WRAD + (size_t)(kb * 20 + nb) * 512, lane), acc);
      int col = nb * 16 + colL;
      float bb = b_rad[col];
#pragma unroll
      for (int r = 0; r < 8; ++r) radS[(r0 + r) * 320 + col] = acc[r] + bb;
    }
    for (int nb = 0; nb < 12; ++nb) {
      v8f acc = {};
#pragma unroll
      for (int kb = 0; kb < 2; ++kb)
        acc = wmma_bf16(al[kb], bfrag(wpack + OFF_WENV + (size_t)(kb * 12 + nb) * 512, lane), acc);
      store_tile(wS, 192, nb * 16, lane, acc);
    }
  }
  __syncthreads();

  // ---- phase 2: gather + Wigner rotate -> XR[9][16][96] bf16 ----
  for (int j = lane; j < 16 * 96; j += 32) {
    int e = j / 96, c = j % 96;
    float v0, v1[3], v2[5];
    if (c < 64) {
      const float* p = nn + (size_t)ecS[e] * 576;
      v0 = p[c];
#pragma unroll
      for (int m = 0; m < 3; ++m) v1[m] = p[64 + c * 3 + m];
#pragma unroll
      for (int m = 0; m < 5; ++m) v2[m] = p[256 + c * 5 + m];
    } else {
      const float* p = edge_features + (size_t)(e0 + e) * 288;
      int ce = c - 64;
      v0 = p[ce];
#pragma unroll
      for (int m = 0; m < 3; ++m) v1[m] = p[32 + ce * 3 + m];
#pragma unroll
      for (int m = 0; m < 5; ++m) v2[m] = p[128 + ce * 5 + m];
    }
    XR[e * 96 + c] = (bf16_t)(v0 * D0S[e]);
#pragma unroll
    for (int m = 0; m < 3; ++m) {
      float s = 0.f;
#pragma unroll
      for (int n = 0; n < 3; ++n) s += D1S[e * 9 + m * 3 + n] * v1[n];
      XR[(1 + m) * 1536 + e * 96 + c] = (bf16_t)s;
    }
#pragma unroll
    for (int m = 0; m < 5; ++m) {
      float s = 0.f;
#pragma unroll
      for (int n = 0; n < 5; ++n) s += D2S[e * 25 + m * 5 + n] * v2[n];
      XR[(4 + m) * 1536 + e * 96 + c] = (bf16_t)s;
    }
  }
  __syncthreads();

  // ---- phase 3: forward GEMMs (A-fragments hoisted to VGPRs once) ----
  {
    v16bf aA[9];
#pragma unroll
    for (int kb = 0; kb < 9; ++kb) {
      const int sl0[9] = {0, 0, 0, 2, 2, 2, 6, 6, 6};   // X0 = [x0r | x1r m=1 | x2r m=2]
      aA[kb] = afrag_lds(XR + sl0[kb] * 1536, 96, (kb % 3) * 32, lane);
    }
    for (int nb = 0; nb < 20; ++nb) {
      v8f acc = {};
#pragma unroll
      for (int kb = 0; kb < 9; ++kb)
        acc = wmma_bf16(aA[kb], bfrag(wpack + OFF_WM0 + (size_t)(kb * 20 + nb) * 512, lane), acc);
      store_tile(Y0S, 320, nb * 16, lane, acc);
    }
  }
  {
    v16bf ap[6], am[6];
#pragma unroll
    for (int kb = 0; kb < 6; ++kb) {
      int sp = (kb < 3) ? 3 : 7, sm = (kb < 3) ? 1 : 5, c0 = (kb % 3) * 32;
      ap[kb] = afrag_lds(XR + sp * 1536, 96, c0, lane);
      am[kb] = afrag_lds(XR + sm * 1536, 96, c0, lane);
    }
    for (int nb = 0; nb < 8; ++nb) {          // Yp1 = Xp1@Wr - Xm1@Wi ; Ym1 = Xp1@Wi + Xm1@Wr
      v8f accP = {}, accM = {};
#pragma unroll
      for (int kb = 0; kb < 6; ++kb) {
        size_t toff = (size_t)(kb * 8 + nb) * 512;
        v16bf br  = bfrag(wpack + OFF_WM1R  + toff, lane);
        v16bf bi  = bfrag(wpack + OFF_WM1I  + toff, lane);
        v16bf bin = bfrag(wpack + OFF_WM1IN + toff, lane);
        accP = wmma_bf16(ap[kb], br,  accP);
        accP = wmma_bf16(am[kb], bin, accP);
        accM = wmma_bf16(ap[kb], bi,  accM);
        accM = wmma_bf16(am[kb], br,  accM);
      }
      store_tile(Yp1S, 128, nb * 16, lane, accP);
      store_tile(Ym1S, 128, nb * 16, lane, accM);
    }
  }
  {
    v16bf ap[3], am[3];
#pragma unroll
    for (int kb = 0; kb < 3; ++kb) {
      ap[kb] = afrag_lds(XR + 8 * 1536, 96, kb * 32, lane);
      am[kb] = afrag_lds(XR + 4 * 1536, 96, kb * 32, lane);
    }
    for (int nb = 0; nb < 4; ++nb) {          // Yp2 / Ym2
      v8f accP = {}, accM = {};
#pragma unroll
      for (int kb = 0; kb < 3; ++kb) {
        size_t toff = (size_t)(kb * 4 + nb) * 512;
        v16bf br  = bfrag(wpack + OFF_WM2R  + toff, lane);
        v16bf bi  = bfrag(wpack + OFF_WM2I  + toff, lane);
        v16bf bin = bfrag(wpack + OFF_WM2IN + toff, lane);
        accP = wmma_bf16(ap[kb], br,  accP);
        accP = wmma_bf16(am[kb], bin, accP);
        accM = wmma_bf16(ap[kb], bi,  accM);
        accM = wmma_bf16(am[kb], br,  accM);
      }
      store_tile(Yp2S, 64, nb * 16, lane, accP);
      store_tile(Ym2S, 64, nb * 16, lane, accM);
    }
  }
  __syncthreads();

  // ---- phase 4: rad scale, back-rotate, silu/sigmoid gate -> MS[9][16][64] bf16 (aliases XR) ----
  bf16_t* MS = XR;
  for (int j = lane; j < 16 * 64; j += 32) {
    int e = j >> 6, c = j & 63;
    float d0 = D0S[e];
    float sci = Y0S[e * 320 + c] * radS[e * 320 + c] * d0;
    float sc  = sci / (1.f + __expf(-sci));
    float g1i = Y0S[e * 320 + 64 + c]  * radS[e * 320 + 64 + c]  * d0;
    float g2i = Y0S[e * 320 + 128 + c] * radS[e * 320 + 128 + c] * d0;
    float g1 = 1.f / (1.f + __expf(-g1i));
    float g2 = 1.f / (1.f + __expf(-g2i));
    float r1 = radS[e * 320 + 192 + c];
    float y1m[3] = { Ym1S[e * 128 + c] * r1, Y0S[e * 320 + 192 + c] * r1, Yp1S[e * 128 + c] * r1 };
    float r2 = radS[e * 320 + 256 + c];
    float y2m[5] = { Ym2S[e * 64 + c] * r2, Ym1S[e * 128 + 64 + c] * r2, Y0S[e * 320 + 256 + c] * r2,
                     Yp1S[e * 128 + 64 + c] * r2, Yp2S[e * 64 + c] * r2 };
    MS[j] = (bf16_t)sc;
#pragma unroll
    for (int n = 0; n < 3; ++n) {
      float s = 0.f;
#pragma unroll
      for (int m = 0; m < 3; ++m) s += D1S[e * 9 + m * 3 + n] * y1m[m];
      MS[(1 + n) * 1024 + j] = (bf16_t)(s * g1);
    }
#pragma unroll
    for (int n = 0; n < 5; ++n) {
      float s = 0.f;
#pragma unroll
      for (int m = 0; m < 5; ++m) s += D2S[e * 25 + m * 5 + n] * y2m[m];
      MS[(4 + n) * 1024 + j] = (bf16_t)(s * g2);
    }
  }
  __syncthreads();

  // ---- phase 5: p GEMMs, env-weight scale, atomic scatter-add ----
  {                                           // p0 = sc@Wp0 + bp0, * w[:, :64]
    v16bf a0[2];
#pragma unroll
    for (int kb = 0; kb < 2; ++kb) a0[kb] = afrag_lds(MS, 64, kb * 32, lane);
    for (int nb = 0; nb < 4; ++nb) {
      v8f acc = {};
#pragma unroll
      for (int kb = 0; kb < 2; ++kb)
        acc = wmma_bf16(a0[kb], bfrag(wpack + OFF_WP0 + (size_t)(kb * 4 + nb) * 512, lane), acc);
      int col = nb * 16 + colL;
      float bb = bp0[col];
#pragma unroll
      for (int r = 0; r < 8; ++r) {
        int e = r0 + r;
        atomic_add_f32(accg + (size_t)ecS[e] * 576 + col, (acc[r] + bb) * wS[e * 192 + col]);
      }
    }
  }
  for (int m = 0; m < 3; ++m) {               // p1[:, d, m]
    v16bf a1[2];
#pragma unroll
    for (int kb = 0; kb < 2; ++kb) a1[kb] = afrag_lds(MS + (1 + m) * 1024, 64, kb * 32, lane);
    for (int nb = 0; nb < 4; ++nb) {
      v8f acc = {};
#pragma unroll
      for (int kb = 0; kb < 2; ++kb)
        acc = wmma_bf16(a1[kb], bfrag(wpack + OFF_WP1 + (size_t)(kb * 4 + nb) * 512, lane), acc);
      int col = nb * 16 + colL;
#pragma unroll
      for (int r = 0; r < 8; ++r) {
        int e = r0 + r;
        atomic_add_f32(accg + (size_t)ecS[e] * 576 + 64 + col * 3 + m,
                       acc[r] * wS[e * 192 + 64 + col]);
      }
    }
  }
  for (int m = 0; m < 5; ++m) {               // p2[:, d, m]
    v16bf a2[2];
#pragma unroll
    for (int kb = 0; kb < 2; ++kb) a2[kb] = afrag_lds(MS + (4 + m) * 1024, 64, kb * 32, lane);
    for (int nb = 0; nb < 4; ++nb) {
      v8f acc = {};
#pragma unroll
      for (int kb = 0; kb < 2; ++kb)
        acc = wmma_bf16(a2[kb], bfrag(wpack + OFF_WP2 + (size_t)(kb * 4 + nb) * 512, lane), acc);
      int col = nb * 16 + colL;
#pragma unroll
      for (int r = 0; r < 8; ++r) {
        int e = r0 + r;
        atomic_add_f32(accg + (size_t)ecS[e] * 576 + 256 + col * 5 + m,
                       acc[r] * wS[e * 192 + 128 + col]);
      }
    }
  }
}

// ---------------- finalize: residual + one-hot channel scaling ----------------
__global__ void finalize_kernel(const float* __restrict__ nf, const float* __restrict__ accg,
                                const int* __restrict__ atom_type,
                                const float* __restrict__ w_oh0, const float* __restrict__ w_oh1,
                                const float* __restrict__ w_oh2, float* __restrict__ out) {
  int idx = blockIdx.x * 256 + threadIdx.x;
  if (idx >= NNODES * 576) return;
  int node = idx / 576, j = idx - node * 576;
  int at = atom_type[node];
  float sh;
  if (j < 64)       sh = A0C * w_oh0[j * 95 + at];
  else if (j < 256) sh = A1C * w_oh1[((j - 64) / 3) * 95 + at];
  else              sh = A2C * w_oh2[((j - 256) / 5) * 95 + at];
  out[idx] = (C_OLD * nf[idx] + C_NEW * NORM8 * accg[idx]) * sh;
}

extern "C" void kernel_launch(void* const* d_in, const int* in_sizes, int n_in,
                              void* d_out, int out_size, void* d_ws, size_t ws_size,
                              hipStream_t stream) {
  const float* node_features = (const float*)d_in[0];
  const float* latents       = (const float*)d_in[1];
  const float* edge_features = (const float*)d_in[2];
  const float* wigner        = (const float*)d_in[5];
  const int*   atom_type     = (const int*)  d_in[7];
  const int*   edge_index    = (const int*)  d_in[8];   // row 0 = source nodes
  const float* ln_w0 = (const float*)d_in[10];
  const float* ln_b0 = (const float*)d_in[11];
  const float* ln_w1 = (const float*)d_in[12];
  const float* ln_w2 = (const float*)d_in[13];
  const float* W_m0  = (const float*)d_in[14];
  const float* W_m1r = (const float*)d_in[15];
  const float* W_m1i = (const float*)d_in[16];
  const float* W_m2r = (const float*)d_in[17];
  const float* W_m2i = (const float*)d_in[18];
  const float* W_rad = (const float*)d_in[19];
  const float* b_rad = (const float*)d_in[20];
  const float* Wp0   = (const float*)d_in[21];
  const float* bp0   = (const float*)d_in[22];
  const float* Wp1   = (const float*)d_in[23];
  const float* Wp2   = (const float*)d_in[24];
  const float* W_env = (const float*)d_in[25];
  const float* w_oh0 = (const float*)d_in[26];
  const float* w_oh1 = (const float*)d_in[27];
  const float* w_oh2 = (const float*)d_in[28];
  float* out = (float*)d_out;

  // workspace: nn (N*576 f32) | acc (N*576 f32) | packed bf16 weights
  float*  nn    = (float*)d_ws;
  float*  accg  = nn + (size_t)NNODES * 576;
  bf16_t* wpack = (bf16_t*)(accg + (size_t)NNODES * 576);

  hipMemsetAsync(accg, 0, (size_t)NNODES * 576 * sizeof(float), stream);

  // pack weights (and negated imaginary parts) into WMMA B-fragment order
  struct PW { const float* w; unsigned off; int K, N; float s; };
  const PW pw[12] = {
    {W_m0,  OFF_WM0,   288, 320,  1.f}, {W_m1r, OFF_WM1R,  192, 128,  1.f},
    {W_m1i, OFF_WM1I,  192, 128,  1.f}, {W_m1i, OFF_WM1IN, 192, 128, -1.f},
    {W_m2r, OFF_WM2R,   96,  64,  1.f}, {W_m2i, OFF_WM2I,   96,  64,  1.f},
    {W_m2i, OFF_WM2IN,  96,  64, -1.f}, {W_rad, OFF_WRAD,   64, 320,  1.f},
    {W_env, OFF_WENV,   64, 192,  1.f}, {Wp0,   OFF_WP0,    64,  64,  1.f},
    {Wp1,   OFF_WP1,    64,  64,  1.f}, {Wp2,   OFF_WP2,    64,  64,  1.f},
  };
  for (int i = 0; i < 12; ++i) {
    int total = pw[i].K * pw[i].N;
    pack_weights_kernel<<<(total + 255) / 256, 256, 0, stream>>>(
        pw[i].w, wpack + pw[i].off, pw[i].K, pw[i].N, pw[i].s);
  }

  node_ln_kernel<<<(NNODES + 7) / 8, 256, 0, stream>>>(node_features, nn, ln_w0, ln_b0, ln_w1, ln_w2);

  const size_t lds_bytes = 109824;
  hipFuncSetAttribute((const void*)edge_kernel,
                      hipFuncAttributeMaxDynamicSharedMemorySize, (int)lds_bytes);
  edge_kernel<<<NEDGES / 16, 32, lds_bytes, stream>>>(
      nn, accg, latents, edge_features, wigner, edge_index, b_rad, bp0, wpack);

  finalize_kernel<<<(NNODES * 576 + 255) / 256, 256, 0, stream>>>(
      node_features, accg, atom_type, w_oh0, w_oh1, w_oh2, out);
}
